// MixVisionTransformer_21792664060513
// MI455X (gfx1250) — compile-verified, matrix operand order checked by reading, not run on hardware
//
#include <hip/hip_runtime.h>

// ---------------------------------------------------------------------------
// Types for CDNA5 WMMA (wave32)
// ---------------------------------------------------------------------------
typedef __bf16 v4bf  __attribute__((ext_vector_type(4)));
typedef __bf16 v8bf  __attribute__((ext_vector_type(8)));
typedef __bf16 v16bf __attribute__((ext_vector_type(16)));
typedef float  v4f   __attribute__((ext_vector_type(4)));
typedef float  v8f   __attribute__((ext_vector_type(8)));

#define DEV __device__ __forceinline__

DEV float siluf(float x)      { return x / (1.f + __expf(-x)); }
DEV float geluf(float x)      { return 0.5f * x * (1.f + erff(x * 0.70710678118654752f)); }
DEV float softplusf(float x)  { return (x > 20.f) ? x : log1pf(__expf(x)); }

// ---------------------------------------------------------------------------
// WMMA GEMM:  Y(M,N) = act( X(M,K) @ W(N,K)^T + bias ) [+= if accum]
//   X row stride = ldx, W row stride = K, Y row stride = ldy.
// Block: 128 threads = 4 waves; block tile 64x64; K tile 64 (2 WMMA K-steps).
// Wave tile 32x32 -> 2x2 WMMA per K-step -> 8 static v_wmma per iteration.
// Staging: fp32 -> bf16, one aligned float4 (global_load_b128) + v4bf
// (ds_store_b64) per thread per 4 elements; single branch per vector.
// All K and ld values in this model are multiples of 4 and all base pointers
// 16B aligned, so vector loads are safe; rows are zero-padded at tile edges.
// Fragment layout per CDNA5 ISA 7.12.2 (16-bit A 16x32 / B 32x16 / f32 C 8 VGPR).
// ---------------------------------------------------------------------------
#define GT_BM 64
#define GT_BN 64
#define GT_BK 64

__global__ __launch_bounds__(128)
void gemm_wmma_kernel(const float* __restrict__ X, int ldx,
                      const float* __restrict__ W,
                      const float* __restrict__ bias,
                      float* __restrict__ Y, int ldy,
                      int M, int N, int K, int act, int accum)
{
    __shared__ __align__(16) __bf16 lA[GT_BM * GT_BK];
    __shared__ __align__(16) __bf16 lB[GT_BN * GT_BK];

    const int tid  = threadIdx.x;
    const int lane = tid & 31;
    const int wave = tid >> 5;
    const int bm = blockIdx.y * GT_BM;
    const int bn = blockIdx.x * GT_BN;
    const int mbase = (wave >> 1) * 32;
    const int nbase = (wave & 1) * 32;
    const int half = lane >> 4;
    const int r16  = lane & 15;

    // staging coordinates: 16 float4-groups per 64-wide row, 8 rows per pass
    const int srow = tid >> 4;        // 0..7
    const int scol = (tid & 15) * 4;  // 0,4,...,60

    v8f acc[2][2];
#pragma unroll
    for (int mi = 0; mi < 2; ++mi)
#pragma unroll
        for (int ni = 0; ni < 2; ++ni)
#pragma unroll
            for (int j = 0; j < 8; ++j) acc[mi][ni][j] = 0.f;

    for (int k0 = 0; k0 < K; k0 += GT_BK) {
        // ---- stage fp32 -> bf16 tiles (vectorized, zero-padded) ----
#pragma unroll
        for (int p = 0; p < 8; ++p) {
            int r  = p * 8 + srow;        // 0..63
            int gk = k0 + scol;
            // A tile
            int gm = bm + r;
            v4f va = {0.f, 0.f, 0.f, 0.f};
            if (gm < M && gk < K) va = *(const v4f*)(X + (size_t)gm * ldx + gk);
            v4bf a4 = {(__bf16)va.x, (__bf16)va.y, (__bf16)va.z, (__bf16)va.w};
            *(v4bf*)(lA + r * GT_BK + scol) = a4;
            // B tile (rows of W)
            int gn = bn + r;
            v4f vb = {0.f, 0.f, 0.f, 0.f};
            if (gn < N && gk < K) vb = *(const v4f*)(W + (size_t)gn * K + gk);
            v4bf b4 = {(__bf16)vb.x, (__bf16)vb.y, (__bf16)vb.z, (__bf16)vb.w};
            *(v4bf*)(lB + r * GT_BK + scol) = b4;
        }
        // Prefetch next K tile (emits global_prefetch_b8 on gfx1250).
        if (k0 + GT_BK < K) {
            int pr = bm + (tid >> 1); if (pr >= M) pr = M - 1;
            __builtin_prefetch(X + (size_t)pr * ldx + (k0 + GT_BK), 0, 1);
            int qr = bn + (tid >> 1); if (qr >= N) qr = N - 1;
            __builtin_prefetch(W + (size_t)qr * K + (k0 + GT_BK), 0, 1);
        }
        __syncthreads();

        // ---- two WMMA K-steps over the 64-wide tile ----
#pragma unroll
        for (int kk = 0; kk < GT_BK; kk += 32) {
            v16bf af[2], bfr[2];
#pragma unroll
            for (int t = 0; t < 2; ++t) {
                int ar = mbase + t * 16 + r16;
                v8bf lo = *(const v8bf*)(lA + ar * GT_BK + kk + half * 8);
                v8bf hi = *(const v8bf*)(lA + ar * GT_BK + kk + 16 + half * 8);
                v16bf v;
#pragma unroll
                for (int i = 0; i < 8; ++i) { v[i] = lo[i]; v[i + 8] = hi[i]; }
                af[t] = v;

                int br = nbase + t * 16 + r16;
                lo = *(const v8bf*)(lB + br * GT_BK + kk + half * 8);
                hi = *(const v8bf*)(lB + br * GT_BK + kk + 16 + half * 8);
#pragma unroll
                for (int i = 0; i < 8; ++i) { v[i] = lo[i]; v[i + 8] = hi[i]; }
                bfr[t] = v;
            }
#pragma unroll
            for (int mi = 0; mi < 2; ++mi)
#pragma unroll
                for (int ni = 0; ni < 2; ++ni)
                    acc[mi][ni] = __builtin_amdgcn_wmma_f32_16x16x32_bf16(
                        false, af[mi], false, bfr[ni], (short)0, acc[mi][ni],
                        false, false);
        }
        __syncthreads();
    }

    // ---- epilogue ----
#pragma unroll
    for (int ni = 0; ni < 2; ++ni) {
        int n = bn + nbase + ni * 16 + r16;
        if (n >= N) continue;
        float bv = bias ? bias[n] : 0.f;
#pragma unroll
        for (int mi = 0; mi < 2; ++mi) {
            int mb = bm + mbase + mi * 16 + half * 8;
#pragma unroll
            for (int j = 0; j < 8; ++j) {
                int m = mb + j;
                if (m < M) {
                    float v = acc[mi][ni][j] + bv;
                    if (act == 1) v = softplusf(v);
                    size_t o = (size_t)m * ldy + n;
                    if (accum) Y[o] += v; else Y[o] = v;
                }
            }
        }
    }
}

// ---------------------------------------------------------------------------
// LayerNorm over last dim: one wave32 per row.
// ---------------------------------------------------------------------------
__global__ __launch_bounds__(256)
void layernorm_kernel(const float* __restrict__ x, const float* __restrict__ g,
                      const float* __restrict__ b, float* __restrict__ y,
                      int R, int C)
{
    int wave = threadIdx.x >> 5, lane = threadIdx.x & 31;
    int row = blockIdx.x * 8 + wave;
    if (row >= R) return;
    const float* xr = x + (size_t)row * C;
    float s = 0.f, s2 = 0.f;
    for (int c = lane; c < C; c += 32) { float v = xr[c]; s += v; s2 += v * v; }
#pragma unroll
    for (int o = 16; o; o >>= 1) { s += __shfl_xor(s, o, 32); s2 += __shfl_xor(s2, o, 32); }
    float inv = 1.f / (float)C;
    float mu = s * inv;
    float var = s2 * inv - mu * mu; if (var < 0.f) var = 0.f;
    float rs = rsqrtf(var + 1e-5f);
    float* yr = y + (size_t)row * C;
    for (int c = lane; c < C; c += 32) yr[c] = (xr[c] - mu) * rs * g[c] + b[c];
}

// ---------------------------------------------------------------------------
// Direct conv2d (patch embed), NCHW, OIHW weights.
// ---------------------------------------------------------------------------
__global__ void conv2d_kernel(const float* __restrict__ in, const float* __restrict__ w,
                              const float* __restrict__ bias, float* __restrict__ out,
                              int B, int Cin, int Hin, int Win_,
                              int Cout, int Hout, int Wout_, int Kk, int stride, int pad)
{
    int idx = blockIdx.x * blockDim.x + threadIdx.x;
    int total = B * Cout * Hout * Wout_;
    if (idx >= total) return;
    int wo = idx % Wout_; int t = idx / Wout_;
    int ho = t % Hout; t /= Hout;
    int co = t % Cout; int b = t / Cout;
    float s = bias[co];
    const float* wp = w + (size_t)co * Cin * Kk * Kk;
    for (int ci = 0; ci < Cin; ++ci)
        for (int kh = 0; kh < Kk; ++kh) {
            int hi = ho * stride - pad + kh;
            if ((unsigned)hi >= (unsigned)Hin) continue;
            for (int kw = 0; kw < Kk; ++kw) {
                int wi = wo * stride - pad + kw;
                if ((unsigned)wi >= (unsigned)Win_) continue;
                s += in[(((size_t)b * Cin + ci) * Hin + hi) * Win_ + wi]
                   * wp[(ci * Kk + kh) * Kk + kw];
            }
        }
    out[idx] = s;
}

// (B,C,L) NCHW -> (B,L,C) tokens
__global__ void tokens_from_nchw_kernel(const float* __restrict__ in, float* __restrict__ out,
                                        int B, int C, int L)
{
    int idx = blockIdx.x * blockDim.x + threadIdx.x;
    int total = B * C * L;
    if (idx >= total) return;
    int l = idx % L; int t = idx / L;
    int c = t % C; int b = t / C;
    out[((size_t)b * L + l) * C + c] = in[idx];
}

// (B,L,C) tokens -> (B,C,L) NCHW
__global__ void nchw_from_tokens_kernel(const float* __restrict__ in, float* __restrict__ out,
                                        int B, int C, int L)
{
    int idx = blockIdx.x * blockDim.x + threadIdx.x;
    int total = B * C * L;
    if (idx >= total) return;
    int l = idx % L; int t = idx / L;
    int c = t % C; int b = t / C;
    out[idx] = in[((size_t)b * L + l) * C + c];
}

// Causal depthwise conv1d (K=4) over L on xi = xz[:, :Di], + SiLU.
__global__ void dwconv1d_silu_kernel(const float* __restrict__ xz, const float* __restrict__ w,
                                     const float* __restrict__ bias, float* __restrict__ xc,
                                     int B, int L, int Di)
{
    int idx = blockIdx.x * blockDim.x + threadIdx.x;
    int total = B * L * Di;
    if (idx >= total) return;
    int ci = idx % Di; int t = idx / Di;
    int l = t % L; int b = t / L;
    int ld = 2 * Di;
    const float* base = xz + (size_t)b * L * ld + ci;
    float s = bias[ci];
#pragma unroll
    for (int j = 0; j < 4; ++j) {
        int ls = l - 3 + j;
        if (ls >= 0) s += base[(size_t)ls * ld] * w[ci * 4 + j];
    }
    xc[idx] = siluf(s);
}

// Selective scan: one thread per (b, channel); 16 states in registers;
// B_t/C_t (32 contiguous floats of dbl row) staged in LDS per step.
__global__ __launch_bounds__(128)
void scan_kernel(const float* __restrict__ u, const float* __restrict__ delta,
                 const float* __restrict__ dbl, int ldbl, int dtr,
                 const float* __restrict__ A_log, const float* __restrict__ Dp,
                 float* __restrict__ y, int B, int L, int Di)
{
    int groups = Di / 128;
    int b  = blockIdx.x / groups;
    int cg = blockIdx.x % groups;
    int ch = cg * 128 + threadIdx.x;
    __shared__ float BC[32];
    float A[16], h[16];
#pragma unroll
    for (int s = 0; s < 16; ++s) { A[s] = -__expf(A_log[ch * 16 + s]); h[s] = 0.f; }
    float Dv = Dp[ch];
    for (int l = 0; l < L; ++l) {
        size_t row = (size_t)b * L + l;
        if (threadIdx.x < 32) BC[threadIdx.x] = dbl[row * ldbl + dtr + threadIdx.x];
        __syncthreads();
        float dt = delta[row * Di + ch];
        float uu = u[row * Di + ch];
        float acc2 = 0.f;
#pragma unroll
        for (int s = 0; s < 16; ++s) {
            h[s] = __expf(dt * A[s]) * h[s] + dt * BC[s] * uu;
            acc2 += h[s] * BC[16 + s];
        }
        y[row * Di + ch] = acc2 + uu * Dv;
        __syncthreads();
    }
}

// y *= silu(z), z = xz[:, Di:2Di]
__global__ void gate_silu_kernel(float* __restrict__ y, const float* __restrict__ xz,
                                 int rows, int Di)
{
    int idx = blockIdx.x * blockDim.x + threadIdx.x;
    int total = rows * Di;
    if (idx >= total) return;
    int ci = idx % Di; int r = idx / Di;
    float z = xz[(size_t)r * 2 * Di + Di + ci];
    y[idx] *= siluf(z);
}

// 3x3 depthwise conv on (B,L,Ch) token layout (spatial HxW), + bias + GELU.
__global__ void dwconv3x3_gelu_kernel(const float* __restrict__ x, const float* __restrict__ w,
                                      const float* __restrict__ bias, float* __restrict__ y,
                                      int B, int L, int Ch, int H, int Wd)
{
    int idx = blockIdx.x * blockDim.x + threadIdx.x;
    int total = B * L * Ch;
    if (idx >= total) return;
    int c = idx % Ch; int t = idx / Ch;
    int l = t % L; int b = t / L;
    int hh = l / Wd, ww = l % Wd;
    float s = bias[c];
#pragma unroll
    for (int dh = 0; dh < 3; ++dh) {
        int hi = hh + dh - 1;
        if ((unsigned)hi >= (unsigned)H) continue;
#pragma unroll
        for (int dw = 0; dw < 3; ++dw) {
            int wi = ww + dw - 1;
            if ((unsigned)wi >= (unsigned)Wd) continue;
            s += x[((size_t)b * L + hi * Wd + wi) * Ch + c] * w[c * 9 + dh * 3 + dw];
        }
    }
    y[idx] = geluf(s);
}

// ---------------------------------------------------------------------------
// Host orchestration
// ---------------------------------------------------------------------------
extern "C" void kernel_launch(void* const* d_in, const int* in_sizes, int n_in,
                              void* d_out, int out_size, void* d_ws, size_t ws_size,
                              hipStream_t stream)
{
    (void)in_sizes; (void)n_in; (void)out_size; (void)ws_size;
    const int DIMS[4] = {64, 128, 256, 512};
    const int KSZ[4]  = {7, 3, 3, 3};
    const int STRD[4] = {4, 2, 2, 2};
    const int DEPTH = 2, Bn = 8;

    int pi = 0;
    const float* xin = (const float*)d_in[pi++];
    auto P = [&]() { return (const float*)d_in[pi++]; };

    float* outp = (float*)d_out;
    size_t out_off = 0;
    const float* cur_in = xin;
    int Hin = 224, Win_ = 224, Cin = 3;

    auto ew = [&](int total) { return dim3((unsigned)((total + 255) / 256)); };

    auto gemm = [&](const float* X, int ldx, const float* W, const float* bias,
                    float* Y, int ldy, int Mm, int Nn, int Kk, int act, int accum) {
        dim3 g((unsigned)((Nn + GT_BN - 1) / GT_BN), (unsigned)((Mm + GT_BM - 1) / GT_BM));
        gemm_wmma_kernel<<<g, 128, 0, stream>>>(X, ldx, W, bias, Y, ldy, Mm, Nn, Kk, act, accum);
    };
    auto lnorm = [&](const float* x, const float* g, const float* b, float* y, int R, int C) {
        layernorm_kernel<<<dim3((unsigned)((R + 7) / 8)), 256, 0, stream>>>(x, g, b, y, R, C);
    };

    for (int si = 0; si < 4; ++si) {
        int d = DIMS[si], k = KSZ[si], st = STRD[si], pad = k / 2;
        int H  = (Hin + 2 * pad - k) / st + 1;
        int Wd = (Win_ + 2 * pad - k) / st + 1;
        int L = H * Wd, M = Bn * L;
        int Di = 2 * d;
        int dtr = (d / 16 > 0) ? d / 16 : 1;
        int hid = 4 * d;                   // hid == 2*Di
        int ldbl = dtr + 32;

        // ---- scratch (bump allocator, reset per stage) ----
        size_t off = 0;
        auto alloc = [&](size_t elems) -> float* {
            float* p = (float*)d_ws + off;
            off += (elems + 255) & ~(size_t)255;
            return p;
        };
        float* tok0 = alloc((size_t)M * d);
        float* tok1 = alloc((size_t)M * d);
        float* lnb  = alloc((size_t)M * d);
        float* xz   = alloc((size_t)M * 2 * Di);   // also reused as y1 (M*hid)
        float* big2 = alloc((size_t)M * 2 * Di);   // xc + delta; later y2 (M*hid)
        float* xc   = big2;
        float* dlt  = big2 + (size_t)M * Di;
        float* dblb = alloc((size_t)M * ldbl);
        float* ybuf = alloc((size_t)M * Di);
        float* cvb  = alloc((size_t)M * d);

        // ---- stage params (setup_inputs insertion order) ----
        const float* pe_w = P(); const float* pe_b = P();
        const float* pe_g = P(); const float* pe_beta = P();
        const float* ng = P();   const float* nb = P();

        // ---- patch embed ----
        conv2d_kernel<<<ew(M * d), 256, 0, stream>>>(cur_in, pe_w, pe_b, cvb,
            Bn, Cin, Hin, Win_, d, H, Wd, k, st, pad);
        tokens_from_nchw_kernel<<<ew(M * d), 256, 0, stream>>>(cvb, lnb, Bn, d, L);
        lnorm(lnb, pe_g, pe_beta, tok0, M, d);

        float* cur = tok0;
        float* alt = tok1;

        for (int bi = 0; bi < DEPTH; ++bi) {
            const float* n1g = P(); const float* n1b = P();
            const float* mng = P(); const float* mnb = P();
            const float* Win = P(); const float* conv_w = P(); const float* conv_b = P();
            const float* Wx  = P(); const float* Wdt = P();    const float* bdt = P();
            const float* A_log = P(); const float* Dp = P();   const float* Wout = P();
            const float* n2g = P(); const float* n2b = P();
            const float* fc1w = P(); const float* fc1b = P();
            const float* dww = P();  const float* dwb = P();
            const float* fc2w = P(); const float* fc2b = P();

            // x = LN(x, n1)
            lnorm(cur, n1g, n1b, alt, M, d);
            { float* t2 = cur; cur = alt; alt = t2; }

            // ---- mamba mix ----
            lnorm(cur, mng, mnb, lnb, M, d);
            gemm(lnb, d, Win, nullptr, xz, 2 * Di, M, 2 * Di, d, 0, 0);
            dwconv1d_silu_kernel<<<ew(M * Di), 256, 0, stream>>>(xz, conv_w, conv_b, xc, Bn, L, Di);
            gemm(xc, Di, Wx, nullptr, dblb, ldbl, M, dtr + 32, Di, 0, 0);
            gemm(dblb, ldbl, Wdt, bdt, dlt, Di, M, Di, dtr, /*softplus*/1, 0);
            scan_kernel<<<dim3((unsigned)(Bn * (Di / 128))), 128, 0, stream>>>(
                xc, dlt, dblb, ldbl, dtr, A_log, Dp, ybuf, Bn, L, Di);
            gate_silu_kernel<<<ew(M * Di), 256, 0, stream>>>(ybuf, xz, M, Di);
            gemm(ybuf, Di, Wout, nullptr, cur, d, M, d, Di, 0, /*accum*/1);

            // ---- MLP ----
            lnorm(cur, n2g, n2b, lnb, M, d);
            gemm(lnb, d, fc1w, fc1b, xz /*y1*/, hid, M, hid, d, 0, 0);
            dwconv3x3_gelu_kernel<<<ew(M * hid), 256, 0, stream>>>(xz, dww, dwb, big2 /*y2*/,
                Bn, L, hid, H, Wd);
            gemm(big2, hid, fc2w, fc2b, cur, d, M, d, hid, 0, /*accum*/1);
        }

        // ---- stage output ----
        lnorm(cur, ng, nb, lnb, M, d);
        nchw_from_tokens_kernel<<<ew(M * d), 256, 0, stream>>>(lnb, outp + out_off, Bn, d, L);

        cur_in = outp + out_off;
        out_off += (size_t)M * d;
        Hin = H; Win_ = Wd; Cin = d;
    }
}
